// SOM_32641751450114
// MI455X (gfx1250) — matrix operand order repaired and checked by reading, not compiled, not run on hardware
//
#include <hip/hip_runtime.h>

typedef __attribute__((ext_vector_type(16))) __bf16 v16bf;
typedef __attribute__((ext_vector_type(8)))  float  v8f;

#define NPTS   65536
#define KUNITS 1024
#define DFEAT  256

// ---------------------------------------------------------------------------
// Kernel 1: convert W to bf16 hi/lo split and compute W2[k] = sum_d W[k,d]^2
// ---------------------------------------------------------------------------
__global__ void som_prep_w(const float* __restrict__ W,
                           unsigned short* __restrict__ Whi,
                           unsigned short* __restrict__ Wlo,
                           float* __restrict__ W2) {
    const int k = blockIdx.x;
    const int d = threadIdx.x;
    float x = W[k * DFEAT + d];
    unsigned u  = __float_as_uint(x);
    unsigned hb = (u + 0x7FFFu + ((u >> 16) & 1u)) >> 16;   // RNE to bf16
    float    hf = __uint_as_float(hb << 16);
    float    lo = x - hf;
    unsigned ul = __float_as_uint(lo);
    unsigned lb = (ul + 0x7FFFu + ((ul >> 16) & 1u)) >> 16;
    Whi[k * DFEAT + d] = (unsigned short)hb;
    Wlo[k * DFEAT + d] = (unsigned short)lb;

    __shared__ float red[256];
    red[d] = x * x;
    __syncthreads();
    for (int s = 128; s > 0; s >>= 1) {
        if (d < s) red[d] += red[d + s];
        __syncthreads();
    }
    if (d == 0) W2[k] = red[0];
}

// ---------------------------------------------------------------------------
// Kernel 2: WMMA distance + argmin with async double-buffered W staging.
//   Each wave owns a 16-point A tile (bf16 hi/lo split, swizzled into the ISA
//   A-matrix VGPR layout, register-resident). W tiles (16 units x 256 dims,
//   hi+lo) are fetched with GLOBAL_LOAD_ASYNC_TO_LDS_B128 directly into the
//   pre-swizzled B-matrix layout, double-buffered so the fetch of tile ut+1
//   overlaps the 24 WMMAs of tile ut. Argmin of  W2[u] - 2*dot  is tracked
//   per lane and reduced with shuffles at the end.
// LDS: per-wave A staging [8][32][16] ushort x2 (8 waves = 128 KB), aliased
//      after A-load with two B double buffers (2 x 16 KB).
// ---------------------------------------------------------------------------
__launch_bounds__(256, 1)
__global__ void som_dist_argmin(const float* __restrict__ X,
                                const unsigned short* __restrict__ Whi,
                                const unsigned short* __restrict__ Wlo,
                                const float* __restrict__ W2,
                                int* __restrict__ bmu) {
    extern __shared__ unsigned short smem[];
    const int t    = threadIdx.x;
    const int w    = t >> 5;          // wave id 0..7
    const int l    = t & 31;          // lane id
    const int half = l >> 4;

    unsigned short* aHi = smem + w * 8192;     // per-wave A staging [8][32][16] x2
    unsigned short* aLo = aHi + 4096;
    // byte offset of dynamic LDS base (low 32 bits of generic ptr = LDS offset)
    const unsigned smemOff = (unsigned)(unsigned long long)(void*)smem;

    const int pbase = blockIdx.x * 128 + w * 16;

    // ---- stage this wave's 16-point A tile: fp32 -> bf16 hi/lo, swizzled ----
    #pragma unroll 4
    for (int i = 0; i < 128; ++i) {
        int e = i * 32 + l;
        int m = e >> 8;               // point row 0..15
        int d = e & 255;              // feature dim
        float x = X[(pbase + m) * DFEAT + d];
        unsigned u  = __float_as_uint(x);
        unsigned hb = (u + 0x7FFFu + ((u >> 16) & 1u)) >> 16;
        float    hf = __uint_as_float(hb << 16);
        float    lo = x - hf;
        unsigned ul = __float_as_uint(lo);
        unsigned lb = (ul + 0x7FFFu + ((ul >> 16) & 1u)) >> 16;
        // A-matrix (16-bit, 16x32) swizzle:  K = 16*(j>>3) + 8*h + (j&7)
        int s  = d >> 5;
        int kl = d & 31;
        int h  = (kl >> 3) & 1;
        int j  = ((kl >> 4) << 3) | (kl & 7);
        int dst = ((s * 32) + (h * 16 + m)) * 16 + j;
        aHi[dst] = (unsigned short)hb;
        aLo[dst] = (unsigned short)lb;
    }
    __syncthreads();

    v16bf ah[8], al[8];
    #pragma unroll
    for (int s = 0; s < 8; ++s) {
        ah[s] = *reinterpret_cast<const v16bf*>(aHi + (s * 32 + l) * 16);
        al[s] = *reinterpret_cast<const v16bf*>(aLo + (s * 32 + l) * 16);
    }
    __syncthreads();   // all waves are register-resident; A LDS is now dead

    // ---- async B staging helper ------------------------------------------
    // B (32x16 bf16) swizzle: element (unit n, dim d) -> [s][h*16+n][8*(kl>>4)+(kl&7)],
    // s=d>>5, kl=d&31, h=(kl>>3)&1. An 8-dim aligned run is contiguous in LDS,
    // so each lane moves 16B with one async B128 transfer (ASYNCcnt).
    auto stageB = [&](unsigned bufOff, int ubase) {
        #pragma unroll
        for (int i = 0; i < 2; ++i) {
            int c  = i * 256 + t;             // 0..511 chunks of 8 dims
            int n  = c >> 5;                  // unit within tile
            int d0 = (c & 31) << 3;           // first dim of chunk
            int s  = d0 >> 5;
            int kl = d0 & 31;
            int h  = (kl >> 3) & 1;
            int j0 = (kl >> 4) << 3;
            unsigned dst = bufOff + (unsigned)((((s * 32) + (h * 16 + n)) * 16 + j0) * 2);
            unsigned long long gHi =
                (unsigned long long)(Whi + (ubase + n) * DFEAT + d0);
            unsigned long long gLo =
                (unsigned long long)(Wlo + (ubase + n) * DFEAT + d0);
            asm volatile("global_load_async_to_lds_b128 %0, %1, off"
                         :: "v"(dst), "v"(gHi) : "memory");
            asm volatile("global_load_async_to_lds_b128 %0, %1, off"
                         :: "v"(dst + 8192u), "v"(gLo) : "memory");
        }
    };

    // prologue: stage tile 0 into buffer 0 (aliases dead A region)
    stageB(smemOff, 0);

    float minv[8];
    int   mini[8];
    #pragma unroll
    for (int r = 0; r < 8; ++r) { minv[r] = 3.4e38f; mini[r] = 0; }

    for (int ut = 0; ut < 64; ++ut) {
        asm volatile("s_wait_asynccnt 0x0" ::: "memory");  // this wave's copies done
        __syncthreads();                                   // everyone's copies done,
                                                           // prev compute on other buf done
        if (ut + 1 < 64)
            stageB(smemOff + (unsigned)(((ut + 1) & 1) * 16384), (ut + 1) * 16);

        const unsigned short* bHi = smem + (ut & 1) * 8192; // 8192 ushorts = 16 KB pair
        const unsigned short* bLo = bHi + 4096;

        v8f c = {0.f, 0.f, 0.f, 0.f, 0.f, 0.f, 0.f, 0.f};
        #pragma unroll
        for (int s = 0; s < 8; ++s) {
            v16bf bh = *reinterpret_cast<const v16bf*>(bHi + (s * 32 + l) * 16);
            v16bf bl = *reinterpret_cast<const v16bf*>(bLo + (s * 32 + l) * 16);
            c = __builtin_amdgcn_wmma_f32_16x16x32_bf16(false, ah[s], false, bh,
                                                        (short)0, c, false, false);
            c = __builtin_amdgcn_wmma_f32_16x16x32_bf16(false, ah[s], false, bl,
                                                        (short)0, c, false, false);
            c = __builtin_amdgcn_wmma_f32_16x16x32_bf16(false, al[s], false, bh,
                                                        (short)0, c, false, false);
        }
        const int   u  = ut * 16 + (l & 15);
        const float w2 = W2[u];
        #pragma unroll
        for (int r = 0; r < 8; ++r) {
            float sc = fmaf(-2.0f, c[r], w2);   // ||W_u||^2 - 2 W_u.X  (X2 const)
            if (sc < minv[r]) { minv[r] = sc; mini[r] = u; }
        }
    }

    // ---- argmin reduction across the 16 lanes of each half (ties -> low idx) ----
    #pragma unroll
    for (int r = 0; r < 8; ++r) {
        float v  = minv[r];
        int   ix = mini[r];
        #pragma unroll
        for (int m = 8; m >= 1; m >>= 1) {
            float ov = __shfl_xor(v, m, 16);
            int   oi = __shfl_xor(ix, m, 16);
            if (ov < v || (ov == v && oi < ix)) { v = ov; ix = oi; }
        }
        if ((l & 15) == r) bmu[pbase + r + half * 8] = ix;  // C layout: M=r / M=r+8
    }
}

// ---------------------------------------------------------------------------
// Kernel 3: scatter  Xsum[bmu[p],:] += X[p,:],  wc[bmu[p]] += 1   (L2 atomics)
// ---------------------------------------------------------------------------
__global__ void som_scatter(const float* __restrict__ X,
                            const int* __restrict__ bmu,
                            float* __restrict__ Xsum,
                            float* __restrict__ wc) {
    const int t = threadIdx.x;
    const int p = blockIdx.x * 8 + (t >> 5);
    const int l = t & 31;
    const int k = bmu[p];
    const float* xr = X + p * DFEAT;
    float* xs = Xsum + k * DFEAT;
    #pragma unroll
    for (int i = 0; i < 8; ++i) {
        int d = l + i * 32;
        atomicAdd(xs + d, xr[d]);
    }
    if (l == 0) atomicAdd(wc + k, 1.0f);
}

// ---------------------------------------------------------------------------
// Kernel 4: S[j] = sum_k wc[k] * U[k,j]
// ---------------------------------------------------------------------------
__global__ void som_s(const float* __restrict__ wc,
                      const float* __restrict__ U,
                      float* __restrict__ S) {
    const int j = blockIdx.x * 256 + threadIdx.x;
    float acc = 0.f;
    for (int k = 0; k < KUNITS; ++k) acc = fmaf(wc[k], U[k * KUNITS + j], acc);
    S[j] = acc;
}

// ---------------------------------------------------------------------------
// Kernel 5: W_new[j,d] = (S[j]>0) ? (sum_k U[k,j]*Xsum[k,d]) / S[j] : W[j,d]
//           and append win_counts.
// ---------------------------------------------------------------------------
__global__ void som_final(const float* __restrict__ U,
                          const float* __restrict__ Xsum,
                          const float* __restrict__ S,
                          const float* __restrict__ W,
                          const float* __restrict__ wc,
                          float* __restrict__ out) {
    const int j = blockIdx.x;
    const int d = threadIdx.x;
    float acc = 0.f;
    for (int k = 0; k < KUNITS; ++k)
        acc = fmaf(U[k * KUNITS + j], Xsum[k * DFEAT + d], acc);
    const float s = S[j];
    out[j * DFEAT + d] = (s > 0.f) ? (acc / s) : W[j * DFEAT + d];
    if (d == 0) out[KUNITS * DFEAT + j] = wc[j];
}

// ---------------------------------------------------------------------------
extern "C" void kernel_launch(void* const* d_in, const int* in_sizes, int n_in,
                              void* d_out, int out_size, void* d_ws, size_t ws_size,
                              hipStream_t stream) {
    const float* X = (const float*)d_in[0];
    const float* W = (const float*)d_in[1];
    const float* U = (const float*)d_in[2];
    float* out = (float*)d_out;

    char* ws = (char*)d_ws;
    unsigned short* Whi = (unsigned short*)(ws);             // 512 KB
    unsigned short* Wlo = (unsigned short*)(ws + 524288);    // 512 KB
    float* W2   = (float*)(ws + 1048576);                    //   4 KB
    float* wc   = (float*)(ws + 1052672);                    //   4 KB
    float* S    = (float*)(ws + 1056768);                    //   4 KB
    int*   bmu  = (int*)  (ws + 1060864);                    // 256 KB
    float* Xsum = (float*)(ws + 1323008);                    //   1 MB

    hipMemsetAsync(wc,   0, KUNITS * sizeof(float), stream);
    hipMemsetAsync(Xsum, 0, KUNITS * DFEAT * sizeof(float), stream);

    som_prep_w<<<KUNITS, 256, 0, stream>>>(W, Whi, Wlo, W2);

    // 512 WGs x 8 waves x 16 points = 65536 points; 128 KB dynamic LDS
    som_dist_argmin<<<NPTS / 128, 256, 131072, stream>>>(X, Whi, Wlo, W2, bmu);

    som_scatter<<<NPTS / 8, 256, 0, stream>>>(X, bmu, Xsum, wc);
    som_s<<<KUNITS / 256, 256, 0, stream>>>(wc, U, S);
    som_final<<<KUNITS, 256, 0, stream>>>(U, Xsum, S, W, wc, out);
}
